// LearnableMolecularPredictor_26946624815836
// MI455X (gfx1250) — compile-verified, matrix operand order checked by reading, not run on hardware
//
#include <hip/hip_runtime.h>
#include <hip/hip_bf16.h>

typedef __attribute__((ext_vector_type(16))) _Float16 v16h;
typedef __attribute__((ext_vector_type(8)))  _Float16 h8;
typedef __attribute__((ext_vector_type(8)))  float    v8f;

#define BB 512
#define TT 2048
#define BN_RSQRT 0.99999500003749980f   // 1/sqrt(1+1e-5)

// -------- WMMA 16-bit operand lane layout (ISA 7.12.2) --------------------
// A (16x32 f16): lanes 0-15 hold row M=lane: elems 0..7 -> K=e, elems 8..15 -> K=e+8(+16)
//                lanes 16-31 hold row M=lane-16: K shifted by +8.
// B (32x16 f16): symmetric, lane (&15) selects column N, same K mapping.
__device__ __forceinline__ int mapK(int e, bool hiHalf) {
    int k = (e < 8) ? e : e + 8;
    return k + (hiHalf ? 8 : 0);
}

// Build a B fragment for weight W stored row-major [out][in]; column 'col',
// K offset 'kofs', zero-padded beyond kmax.
__device__ __forceinline__ v16h make_bfrag(const float* __restrict__ W, int in_dim,
                                           int col, int kofs, bool hiHalf, int kmax) {
    v16h b;
#pragma unroll
    for (int e = 0; e < 16; ++e) {
        int k = kofs + mapK(e, hiHalf);
        b[e] = (k < kmax) ? (_Float16)W[col * in_dim + k] : (_Float16)0.f;
    }
    return b;
}

__device__ __forceinline__ v8f wmma_f16(v16h a, v16h b, v8f c) {
    return __builtin_amdgcn_wmma_f32_16x16x32_f16(false, a, false, b, (short)0, c, false, false);
}

// ============================ Kernel 1 =====================================
// Per-tile pipeline: feats(8) -> h(64) relu+bn -> tf(32) relu -> att(16) tanh -> score
__global__ __launch_bounds__(256)
void k1_tiles(const float* __restrict__ tp,
              const float* __restrict__ feW1, const float* __restrict__ feb1,
              const float* __restrict__ feg,  const float* __restrict__ febe,
              const float* __restrict__ feW2, const float* __restrict__ feb2,
              const float* __restrict__ atW1, const float* __restrict__ atb1,
              const float* __restrict__ atW2, const float* __restrict__ atb2,
              _Float16* __restrict__ tfg, float* __restrict__ scores,
              float* __restrict__ sums)
{
    const int b     = blockIdx.x;
    const int chunk = blockIdx.y;
    const int wv    = threadIdx.x >> 5;
    const int lane  = threadIdx.x & 31;
    const bool hiH  = lane >= 16;
    const int  col  = lane & 15;
    const int  kb   = hiH ? 8 : 0;

    __shared__ __attribute__((aligned(16))) _Float16 hbuf [8][16 * 64];
    __shared__ __attribute__((aligned(16))) _Float16 tfbuf[8][16 * 32];

    // ---- hoisted weight fragments & per-lane scalars ----
    v16h bW1[4];
#pragma unroll
    for (int j = 0; j < 4; ++j) bW1[j] = make_bfrag(feW1, 8, j * 16 + col, 0, hiH, 8);
    v16h bW2[2][2];
#pragma unroll
    for (int c = 0; c < 2; ++c)
#pragma unroll
        for (int j = 0; j < 2; ++j) bW2[c][j] = make_bfrag(feW2, 64, j * 16 + col, 32 * c, hiH, 64);
    v16h bA1 = make_bfrag(atW1, 32, col, 0, hiH, 32);

    float b1v[4], gsv[4], bev[4];
#pragma unroll
    for (int j = 0; j < 4; ++j) {
        b1v[j] = feb1[j * 16 + col];
        gsv[j] = feg [j * 16 + col] * (float)BN_RSQRT;
        bev[j] = febe[j * 16 + col];
    }
    float b2v[2];
#pragma unroll
    for (int j = 0; j < 2; ++j) b2v[j] = feb2[j * 16 + col];
    const float ab1v = atb1[col];
    const float aw2v = atW2[col];
    const float ab2v = atb2[0];

    float s_tu = 0.f, s_im = 0.f, s_st = 0.f;

    for (int it = 0; it < 2; ++it) {
        const int t0 = chunk * 256 + (it * 8 + wv) * 16;   // 16-row tile base

        // ---- per-lane feature computation (rows live in lanes 0-15) ----
        v16h afeat;
#pragma unroll
        for (int e = 0; e < 16; ++e) afeat[e] = (_Float16)0.f;
        if (!hiH) {
            const float* p = tp + (((size_t)b * TT) + t0 + col) * 8;
            float q[8];
#pragma unroll
            for (int i = 0; i < 8; ++i) q[i] = p[i];
            float tumor   = q[0] + 0.5f * q[2];
            float immune  = q[3] + 0.3f * q[5];
            float stromal = q[1] + 0.5f * q[2];
            float ent = 0.f, mx = q[0];
#pragma unroll
            for (int i = 0; i < 8; ++i) {
                ent -= q[i] * __logf(q[i] + 1e-8f);
                mx   = fmaxf(mx, q[i]);
            }
            float f[8] = { tumor, immune, stromal,
                           tumor  / (immune  + 0.01f),
                           tumor  / (stromal + 0.01f),
                           immune / (stromal + 0.01f),
                           ent, mx };
            s_tu += tumor; s_im += immune; s_st += stromal;
#pragma unroll
            for (int e = 0; e < 8; ++e) afeat[e] = (_Float16)f[e];
        }

        // ---- GEMM1: feats[16x8] x W1^T[8x64] -> h, relu+bn, to LDS f16 ----
#pragma unroll
        for (int j = 0; j < 4; ++j) {
            v8f c;
#pragma unroll
            for (int r = 0; r < 8; ++r) c[r] = b1v[j];
            c = wmma_f16(afeat, bW1[j], c);
            const int n = j * 16 + col;
#pragma unroll
            for (int r = 0; r < 8; ++r) {
                float v = fmaxf(c[r], 0.f) * gsv[j] + bev[j];
                int   m = hiH ? r + 8 : r;
                hbuf[wv][m * 64 + n] = (_Float16)v;
            }
        }
        asm volatile("s_wait_dscnt 0x0" ::: "memory");

        // ---- GEMM2: h[16x64] x W2^T[64x32] -> tf, relu, to LDS + global ----
        v8f c2[2];
#pragma unroll
        for (int j = 0; j < 2; ++j)
#pragma unroll
            for (int r = 0; r < 8; ++r) c2[j][r] = b2v[j];
#pragma unroll
        for (int c = 0; c < 2; ++c) {
            h8 lo8 = *(const h8*)&hbuf[wv][col * 64 + 32 * c + kb];
            h8 hi8 = *(const h8*)&hbuf[wv][col * 64 + 32 * c + kb + 16];
            v16h a2 = __builtin_shufflevector(lo8, hi8, 0,1,2,3,4,5,6,7,8,9,10,11,12,13,14,15);
#pragma unroll
            for (int j = 0; j < 2; ++j) c2[j] = wmma_f16(a2, bW2[c][j], c2[j]);
        }
#pragma unroll
        for (int j = 0; j < 2; ++j) {
            const int n = j * 16 + col;
#pragma unroll
            for (int r = 0; r < 8; ++r) {
                float v = fmaxf(c2[j][r], 0.f);
                int   m = hiH ? r + 8 : r;
                tfbuf[wv][m * 32 + n] = (_Float16)v;
            }
        }
        asm volatile("s_wait_dscnt 0x0" ::: "memory");

        {   // coalesced tile_feat writeback (f16, 16 halves per lane)
            _Float16* dst = tfg + ((size_t)b * TT + t0) * 32;
            h8 v0 = *(const h8*)&tfbuf[wv][lane * 16];
            h8 v1 = *(const h8*)&tfbuf[wv][lane * 16 + 8];
            *(h8*)(dst + lane * 16)     = v0;
            *(h8*)(dst + lane * 16 + 8) = v1;
        }

        // ---- GEMM3: tf[16x32] x aW1^T[32x16] -> tanh -> dot aW2 -> score ----
        h8 lo8 = *(const h8*)&tfbuf[wv][col * 32 + kb];
        h8 hi8 = *(const h8*)&tfbuf[wv][col * 32 + kb + 16];
        v16h a3 = __builtin_shufflevector(lo8, hi8, 0,1,2,3,4,5,6,7,8,9,10,11,12,13,14,15);
        v8f c3;
#pragma unroll
        for (int r = 0; r < 8; ++r) c3[r] = ab1v;
        c3 = wmma_f16(a3, bA1, c3);

        float sc[8];
#pragma unroll
        for (int r = 0; r < 8; ++r) {
            float v = tanhf(c3[r]) * aw2v;          // per-column contribution
#pragma unroll
            for (int m = 8; m >= 1; m >>= 1) v += __shfl_xor(v, m);  // sum 16 cols
            sc[r] = v;
        }
        if (col == 0) {                              // lanes 0 (rows 0-7), 16 (rows 8-15)
            float* sp = scores + (size_t)b * TT + t0 + (hiH ? 8 : 0);
#pragma unroll
            for (int r = 0; r < 8; ++r) sp[r] = sc[r] + ab2v;
        }
    }

    // ---- per-batch tumor/immune/stromal sums ----
#pragma unroll
    for (int m = 16; m >= 1; m >>= 1) {
        s_tu += __shfl_xor(s_tu, m);
        s_im += __shfl_xor(s_im, m);
        s_st += __shfl_xor(s_st, m);
    }
    if (lane == 0) {
        atomicAdd(&sums[b * 4 + 0], s_tu);
        atomicAdd(&sums[b * 4 + 1], s_im);
        atomicAdd(&sums[b * 4 + 2], s_st);
    }
}

// ============================ Kernel 2 =====================================
// Per-batch softmax over T and weighted aggregation of tile_feat -> agg[32]
__global__ __launch_bounds__(256)
void k2_softmax_agg(const float* __restrict__ scores, const _Float16* __restrict__ tfg,
                    float* __restrict__ agg)
{
    const int b = blockIdx.x, tid = threadIdx.x;
    __shared__ float sw[TT];
    __shared__ float red[256];
    const float* sb = scores + (size_t)b * TT;

    float mx = -3.402823e38f;
    for (int t = tid; t < TT; t += 256) mx = fmaxf(mx, sb[t]);
    red[tid] = mx; __syncthreads();
    for (int o = 128; o > 0; o >>= 1) {
        if (tid < o) red[tid] = fmaxf(red[tid], red[tid + o]);
        __syncthreads();
    }
    mx = red[0]; __syncthreads();

    float se = 0.f;
    for (int t = tid; t < TT; t += 256) {
        float e = __expf(sb[t] - mx);
        sw[t] = e; se += e;
    }
    red[tid] = se; __syncthreads();
    for (int o = 128; o > 0; o >>= 1) {
        if (tid < o) red[tid] += red[tid + o];
        __syncthreads();
    }
    const float Z = red[0]; __syncthreads();

    const int n = tid & 31, g = tid >> 5;         // 8 t-groups x 32 features
    const _Float16* tb = tfg + (size_t)b * TT * 32;
    float acc = 0.f;
    for (int t = g; t < TT; t += 8) acc += sw[t] * (float)tb[(size_t)t * 32 + n];
    red[tid] = acc; __syncthreads();
    if (tid < 32) {
        float a = 0.f;
#pragma unroll
        for (int gg = 0; gg < 8; ++gg) a += red[gg * 32 + tid];
        agg[b * 32 + tid] = a / Z;
    }
}

// ============================ Kernel 3 =====================================
struct HeadP { const float *W1, *b1, *g, *be, *W2, *b2, *W3, *b3; };

__global__ __launch_bounds__(256)
void k3_heads(const float* __restrict__ agg, const float* __restrict__ sums,
              HeadP p1, HeadP p2, HeadP p3, const float* __restrict__ cw,
              float* __restrict__ out)
{
    const int b = blockIdx.x * blockDim.x + threadIdx.x;
    if (b >= BB) return;
    float ag[32];
#pragma unroll
    for (int i = 0; i < 32; ++i) ag[i] = agg[b * 32 + i];

    auto head = [&](const HeadP& P) -> float {
        float h1[64];
        for (int o = 0; o < 64; ++o) {
            float v = P.b1[o];
            for (int i = 0; i < 32; ++i) v += P.W1[o * 32 + i] * ag[i];
            h1[o] = fmaxf(v, 0.f) * (P.g[o] * (float)BN_RSQRT) + P.be[o];
        }
        float r = P.b3[0];
        for (int o = 0; o < 32; ++o) {
            float v = P.b2[o];
            for (int i = 0; i < 64; ++i) v += P.W2[o * 64 + i] * h1[i];
            r += P.W3[o] * fmaxf(v, 0.f);
        }
        return r;
    };

    float s1 = head(p1), s2 = head(p2), s3 = head(p3);
    const float inv = 1.f / (float)TT;
    float tm = sums[b * 4 + 0] * inv;
    float im = sums[b * 4 + 1] * inv;
    float st = sums[b * 4 + 2] * inv;
    s2 += im * cw[0];
    s3 += st * cw[1];
    s1 += tm * (1.f - im) * cw[2];
    float m  = fmaxf(s1, fmaxf(s2, s3));
    float e1 = __expf(s1 - m), e2 = __expf(s2 - m), e3 = __expf(s3 - m);
    float z  = e1 + e2 + e3;
    out[b * 3 + 0] = e1 / z;
    out[b * 3 + 1] = e2 / z;
    out[b * 3 + 2] = e3 / z;
}

// ============================ Launch =======================================
extern "C" void kernel_launch(void* const* d_in, const int* in_sizes, int n_in,
                              void* d_out, int out_size, void* d_ws, size_t ws_size,
                              hipStream_t stream) {
    const float* tp   = (const float*)d_in[0];
    const float* feW1 = (const float*)d_in[1];
    const float* feb1 = (const float*)d_in[2];
    const float* feg  = (const float*)d_in[3];
    const float* febe = (const float*)d_in[4];
    const float* feW2 = (const float*)d_in[5];
    const float* feb2 = (const float*)d_in[6];
    const float* atW1 = (const float*)d_in[7];
    const float* atb1 = (const float*)d_in[8];
    const float* atW2 = (const float*)d_in[9];
    const float* atb2 = (const float*)d_in[10];
    HeadP p1{ (const float*)d_in[11], (const float*)d_in[12], (const float*)d_in[13],
              (const float*)d_in[14], (const float*)d_in[15], (const float*)d_in[16],
              (const float*)d_in[17], (const float*)d_in[18] };
    HeadP p2{ (const float*)d_in[19], (const float*)d_in[20], (const float*)d_in[21],
              (const float*)d_in[22], (const float*)d_in[23], (const float*)d_in[24],
              (const float*)d_in[25], (const float*)d_in[26] };
    HeadP p3{ (const float*)d_in[27], (const float*)d_in[28], (const float*)d_in[29],
              (const float*)d_in[30], (const float*)d_in[31], (const float*)d_in[32],
              (const float*)d_in[33], (const float*)d_in[34] };
    const float* cw = (const float*)d_in[35];

    // workspace layout (bytes): tile_feat f16 | scores f32 | sums f32 | agg f32
    char* ws = (char*)d_ws;
    _Float16* tfg    = (_Float16*)(ws);                               // 64 MB
    float*    scores = (float*)(ws + (size_t)67108864);               //  4 MB
    float*    sums   = (float*)(ws + (size_t)71303168);               //  8 KB
    float*    agg    = (float*)(ws + (size_t)71311360);               // 64 KB

    hipMemsetAsync(sums, 0, BB * 4 * sizeof(float), stream);

    k1_tiles<<<dim3(BB, 8), 256, 0, stream>>>(tp, feW1, feb1, feg, febe, feW2, feb2,
                                              atW1, atb1, atW2, atb2,
                                              tfg, scores, sums);
    k2_softmax_agg<<<BB, 256, 0, stream>>>(scores, tfg, agg);
    k3_heads<<<2, 256, 0, stream>>>(agg, sums, p1, p2, p3, cw, (float*)d_out);
}
